// PseudoTokenGridEncoder_86251533238894
// MI455X (gfx1250) — matrix operand order repaired and checked by reading, not compiled, not run on hardware
//
#include <hip/hip_runtime.h>

// ---------------------------------------------------------------------------
// Problem constants (from reference)
// ---------------------------------------------------------------------------
#define BB 8
#define NN 16384
#define EE 256
#define GG 32
#define SS (GG * GG)          // 1024
#define MAXP 64
#define NPTS (BB * NN)        // 131072
#define NCELL (BB * SS)       // 8192

typedef __attribute__((ext_vector_type(16))) __bf16 v16bf;
typedef __attribute__((ext_vector_type(8)))  float  v8f;
typedef __attribute__((ext_vector_type(4)))  unsigned int v4u;
typedef __attribute__((ext_vector_type(4)))  unsigned int u32x4;
typedef __attribute__((ext_vector_type(8)))  int i32x8;
typedef __attribute__((ext_vector_type(4)))  int i32x4;

union FragBF { v16bf v; v4u u[2]; };

#define HAS_TDM __has_builtin(__builtin_amdgcn_tensor_load_to_lds)

// f32 -> bf16 bits, round-to-nearest-even
__device__ __forceinline__ unsigned f2bf(float f) {
    unsigned u = __float_as_uint(f);
    unsigned r = u + 0x7FFFu + ((u >> 16) & 1u);
    return r >> 16;
}

// ---------------------------------------------------------------------------
// 0) f32 -> bf16 bulk convert (8 elements / thread, b128 stores)
// ---------------------------------------------------------------------------
__global__ __launch_bounds__(256) void convert_bf16_kernel(const float* __restrict__ src,
                                                           unsigned short* __restrict__ dst) {
    int gid = blockIdx.x * 256 + threadIdx.x;       // one 8-element chunk per thread
    const float4* s = (const float4*)(src + (size_t)gid * 8);
    float4 a = s[0], b = s[1];
    v4u p;
    p[0] = f2bf(a.x) | (f2bf(a.y) << 16);
    p[1] = f2bf(a.z) | (f2bf(a.w) << 16);
    p[2] = f2bf(b.x) | (f2bf(b.y) << 16);
    p[3] = f2bf(b.z) | (f2bf(b.w) << 16);
    *(v4u*)(dst + (size_t)gid * 8) = p;
}

// Transposed bf16 weights: wT[n][k] = bf16(w[k][n]).  <<<256,256>>>
__global__ __launch_bounds__(256) void transpose_w_kernel(const float* __restrict__ w,
                                                          unsigned short* __restrict__ wT) {
    int n = blockIdx.x, k = threadIdx.x;
    wT[n * EE + k] = (unsigned short)f2bf(w[k * EE + n]);
}

// ---------------------------------------------------------------------------
// 1) Binning: per point -> cell, slot via atomics
// ---------------------------------------------------------------------------
__global__ __launch_bounds__(256) void bin_kernel(const float* __restrict__ x,
                                                  int* __restrict__ counts,
                                                  int* __restrict__ cellPts) {
    int gp = blockIdx.x * 256 + threadIdx.x;
    float x0 = x[gp * 2 + 0];
    float x1 = x[gp * 2 + 1];
    int i0 = (int)floorf(x0 * (float)(GG - 1) + 0.5f);
    int i1 = (int)floorf(x1 * (float)(GG - 1) + 0.5f);
    i0 = min(max(i0, 0), GG - 1);
    i1 = min(max(i1, 0), GG - 1);
    int b = gp >> 14;
    int cell = b * SS + i0 * GG + i1;
    int slot = atomicAdd(&counts[cell], 1);
    if (slot < MAXP) cellPts[cell * MAXP + slot] = gp;
}

// ---------------------------------------------------------------------------
// 2) bf16 WMMA GEMM: C[M,256] (f32) = A[M,256] (bf16) @ WT[256,256]^T (bf16,[n][k])
//    grid = (2, M/128), block = 256 (8 waves). Workgroup tile 128x128.
//    Tiles staged via TDM (double-buffered) when available.
// ---------------------------------------------------------------------------
#if HAS_TDM
__device__ __forceinline__ void tdm_load_tile(const unsigned short* base, int rows,
                                              int row0, int k0, unsigned ldsoff) {
    unsigned long long ga =
        (unsigned long long)(uintptr_t)base + ((unsigned long long)row0 * EE + (unsigned)k0) * 2ull;
    u32x4 g0;
    g0[0] = 1u;                                   // count=1, user descriptor
    g0[1] = ldsoff;                               // LDS byte address
    g0[2] = (unsigned)ga;                         // global addr [31:0]
    g0[3] = (unsigned)((ga >> 32) & 0x01FFFFFFull) | (2u << 30);  // addr[56:32] | type=2
    i32x8 g1;
    g1[0] = (1 << 16);                            // data_size = 1 (2 bytes), no mask/pad
    g1[1] = (EE << 16);                           // tensor_dim0[15:0] = 256
    g1[2] = (int)(((unsigned)rows & 0xFFFFu) << 16);              // tensor_dim1 lo
    g1[3] = (int)((((unsigned)rows >> 16) & 0xFFFFu) | (32u << 16)); // dim1 hi | tile_dim0=32
    g1[4] = 128;                                  // tile_dim1 = 128
    g1[5] = EE;                                   // tensor_dim0_stride = 256 elements
    g1[6] = 0;
    g1[7] = 0;
    i32x4 gz4 = {0, 0, 0, 0};                     // 2-D tensor: groups 2/3 unused
    i32x8 gz8 = {0, 0, 0, 0, 0, 0, 0, 0};
    __builtin_amdgcn_tensor_load_to_lds(g0, g1, gz4, gz4, gz8, 0);
}
#endif

__device__ __forceinline__ void wmma_step(const unsigned short* tA, const unsigned short* tW,
                                          v8f acc[8], int rowA, int cA0, int colB, int ckB) {
    FragBF a;
    a.u[0] = *(const v4u*)(tA + rowA * 32 + cA0);
    a.u[1] = *(const v4u*)(tA + rowA * 32 + cA0 + 16);
    FragBF b[8];
#pragma unroll
    for (int j = 0; j < 8; ++j) {
        const unsigned short* pb = tW + (j * 16 + colB) * 32 + ckB;
        b[j].u[0] = *(const v4u*)(pb);
        b[j].u[1] = *(const v4u*)(pb + 16);
    }
#pragma unroll
    for (int j = 0; j < 8; ++j) {
        acc[j] = __builtin_amdgcn_wmma_f32_16x16x32_bf16(
            false, a.v, false, b[j].v, (short)0, acc[j], false, false);
    }
}

__global__ __launch_bounds__(256) void gemm_bf16_wmma(const unsigned short* __restrict__ A,
                                                      const unsigned short* __restrict__ WT,
                                                      float* __restrict__ C, int M) {
    __shared__ __align__(16) unsigned short sm[2][2][128 * 32];  // [buf][A/W][m|n][k]

    const int tid  = threadIdx.x;
    const int wave = tid >> 5;
    const int lane = tid & 31;
    const int m0 = blockIdx.y * 128;
    const int n0 = blockIdx.x * 128;
    const int m0w = wave * 16;

    const v8f vzero = {0.f, 0.f, 0.f, 0.f, 0.f, 0.f, 0.f, 0.f};
    v8f acc[8];
#pragma unroll
    for (int j = 0; j < 8; ++j) acc[j] = vzero;

    // 16-bit A 16x32 layout: lanes 0-15 row=lane, K {0..7}&{16..23};
    //                        lanes16-31 row=lane-16, K {8..15}&{24..31}
    const int rowA = m0w + (lane & 15);
    const int cA0  = (lane & 16) ? 8 : 0;
    // 16-bit B 32x16 layout: lanes 0-15 col=lane K {0..7}&{16..23}; lanes16-31 col=lane-16 K {8..15}&{24..31}
    const int colB = lane & 15;
    const int ckB  = (lane & 16) ? 8 : 0;

#if HAS_TDM
    if (tid < 32) {  // wave 0 drives the Tensor Data Mover
        tdm_load_tile(A,  M,  m0, 0, (unsigned)(uintptr_t)&sm[0][0][0]);
        tdm_load_tile(WT, EE, n0, 0, (unsigned)(uintptr_t)&sm[0][1][0]);
    }
#pragma unroll 1
    for (int kt = 0; kt < 8; ++kt) {
        if (tid < 32) {
            if (kt < 7) {  // prefetch next stage into the other buffer
                tdm_load_tile(A,  M,  m0, (kt + 1) * 32, (unsigned)(uintptr_t)&sm[(kt + 1) & 1][0][0]);
                tdm_load_tile(WT, EE, n0, (kt + 1) * 32, (unsigned)(uintptr_t)&sm[(kt + 1) & 1][1][0]);
                __builtin_amdgcn_s_wait_tensorcnt(2);  // stage kt resident (in-order)
            } else {
                __builtin_amdgcn_s_wait_tensorcnt(0);
            }
        }
        __syncthreads();
        wmma_step(sm[kt & 1][0], sm[kt & 1][1], acc, rowA, cA0, colB, ckB);
        __syncthreads();
    }
#else
#pragma unroll 1
    for (int kt = 0; kt < 8; ++kt) {
        const int k0 = kt * 32;
        __syncthreads();
        // vectorized b128 copy: 512 chunks of 8 bf16, 2 per thread
#pragma unroll
        for (int c = tid; c < 512; c += 256) {
            int row = c >> 2, ko = (c & 3) * 8;
            *(v4u*)(&sm[0][0][c * 8]) = *(const v4u*)(A + (size_t)(m0 + row) * EE + k0 + ko);
            *(v4u*)(&sm[0][1][c * 8]) = *(const v4u*)(WT + (size_t)(n0 + row) * EE + k0 + ko);
        }
        __syncthreads();
        wmma_step(sm[0][0], sm[0][1], acc, rowA, cA0, colB, ckB);
    }
#endif

    // D layout: VGPR i -> row m0w+i (lanes 0-15) / m0w+8+i (lanes 16-31), col = lane&15
    const int rbase = m0 + m0w + ((lane & 16) ? 8 : 0);
    const int cbase = n0 + (lane & 15);
#pragma unroll
    for (int j = 0; j < 8; ++j) {
#pragma unroll
        for (int i = 0; i < 8; ++i) {
            C[(size_t)(rbase + i) * EE + (cbase + j * 16)] = acc[j][i];
        }
    }
}

// ---------------------------------------------------------------------------
// 3) Per-cell online-softmax attention. One wave per cell; bf16 output.
// ---------------------------------------------------------------------------
__global__ __launch_bounds__(256) void attn_kernel(const float* __restrict__ q_lat,
                                                   const float* __restrict__ k_lat,
                                                   const float* __restrict__ v_lat,
                                                   const float* __restrict__ k_pts,
                                                   const float* __restrict__ v_pts,
                                                   const int* __restrict__ counts,
                                                   const int* __restrict__ cellPts,
                                                   unsigned short* __restrict__ outb) {
    const int wave = threadIdx.x >> 5;
    const int lane = threadIdx.x & 31;
    const int cell = blockIdx.x * 8 + wave;
    const int s = cell & (SS - 1);
    const int e0 = lane * 8;
    const float scale = 0.1767766952966369f;     // 1/sqrt(32)

    const float4* qp = (const float4*)(q_lat + (size_t)s * EE + e0);
    float4 qa = qp[0], qb = qp[1];

    const float4* kg = (const float4*)(k_lat + (size_t)s * EE + e0);
    const float4* vg = (const float4*)(v_lat + (size_t)s * EE + e0);
    float4 ka = kg[0], kb = kg[1];
    float4 va = vg[0], vb = vg[1];

    float d = qa.x * ka.x + qa.y * ka.y + qa.z * ka.z + qa.w * ka.w +
              qb.x * kb.x + qb.y * kb.y + qb.z * kb.z + qb.w * kb.w;
    d += __shfl_xor(d, 1, 32);
    d += __shfl_xor(d, 2, 32);

    float m = d * scale;
    float ssum = 1.0f;
    float o[8] = {va.x, va.y, va.z, va.w, vb.x, vb.y, vb.z, vb.w};

    int cnt = counts[cell];
    cnt = (cnt > MAXP) ? MAXP : cnt;

    for (int j = 0; j < cnt; ++j) {
        const int p = cellPts[cell * MAXP + j];
        const float4* kp = (const float4*)(k_pts + (size_t)p * EE + e0);
        const float4* vp = (const float4*)(v_pts + (size_t)p * EE + e0);
        float4 k0 = kp[0], k1 = kp[1];
        float4 v0 = vp[0], v1 = vp[1];

        float dd = qa.x * k0.x + qa.y * k0.y + qa.z * k0.z + qa.w * k0.w +
                   qb.x * k1.x + qb.y * k1.y + qb.z * k1.z + qb.w * k1.w;
        dd += __shfl_xor(dd, 1, 32);
        dd += __shfl_xor(dd, 2, 32);
        float logit = dd * scale;

        float mn = fmaxf(m, logit);
        float corr = __expf(m - mn);
        float w = __expf(logit - mn);
        ssum = ssum * corr + w;
        o[0] = o[0] * corr + w * v0.x;
        o[1] = o[1] * corr + w * v0.y;
        o[2] = o[2] * corr + w * v0.z;
        o[3] = o[3] * corr + w * v0.w;
        o[4] = o[4] * corr + w * v1.x;
        o[5] = o[5] * corr + w * v1.y;
        o[6] = o[6] * corr + w * v1.z;
        o[7] = o[7] * corr + w * v1.w;
        m = mn;
    }

    const float inv = 1.0f / ssum;
    v4u p;
    p[0] = f2bf(o[0] * inv) | (f2bf(o[1] * inv) << 16);
    p[1] = f2bf(o[2] * inv) | (f2bf(o[3] * inv) << 16);
    p[2] = f2bf(o[4] * inv) | (f2bf(o[5] * inv) << 16);
    p[3] = f2bf(o[6] * inv) | (f2bf(o[7] * inv) << 16);
    *(v4u*)(outb + (size_t)cell * EE + e0) = p;
}

// ---------------------------------------------------------------------------
// 4) x_grid broadcast
// ---------------------------------------------------------------------------
__global__ __launch_bounds__(256) void xgrid_kernel(const float* __restrict__ grid,
                                                    float* __restrict__ out) {
    int t = blockIdx.x * 256 + threadIdx.x;
    out[t] = grid[t & (SS * 2 - 1)];
}

// ---------------------------------------------------------------------------
// Launch
// ---------------------------------------------------------------------------
extern "C" void kernel_launch(void* const* d_in, const int* in_sizes, int n_in,
                              void* d_out, int out_size, void* d_ws, size_t ws_size,
                              hipStream_t stream) {
    const float* x    = (const float*)d_in[0];
    const float* z    = (const float*)d_in[1];
    const float* grid = (const float*)d_in[2];
    const float* lat  = (const float*)d_in[3];
    const float* wq   = (const float*)d_in[4];
    const float* wk   = (const float*)d_in[5];
    const float* wv   = (const float*)d_in[6];
    const float* wo   = (const float*)d_in[7];

    float* out = (float*)d_out;

    char* base = (char*)d_ws;
    size_t off = 0;
    int*   counts  = (int*)(base + off);            off += (size_t)NCELL * 4;
    int*   cellPts = (int*)(base + off);            off += (size_t)NCELL * MAXP * 4;
    unsigned short* zb   = (unsigned short*)(base + off); off += (size_t)NPTS * EE * 2;
    unsigned short* latb = (unsigned short*)(base + off); off += (size_t)SS * EE * 2;
    unsigned short* wqT  = (unsigned short*)(base + off); off += (size_t)EE * EE * 2;
    unsigned short* wkT  = (unsigned short*)(base + off); off += (size_t)EE * EE * 2;
    unsigned short* wvT  = (unsigned short*)(base + off); off += (size_t)EE * EE * 2;
    unsigned short* woT  = (unsigned short*)(base + off); off += (size_t)EE * EE * 2;
    float* k_pts = (float*)(base + off);            off += (size_t)NPTS * EE * 4;
    float* v_pts = (float*)(base + off);            off += (size_t)NPTS * EE * 4;
    float* q_lat = (float*)(base + off);            off += (size_t)SS * EE * 4;
    float* k_lat = (float*)(base + off);            off += (size_t)SS * EE * 4;
    float* v_lat = (float*)(base + off);            off += (size_t)SS * EE * 4;
    unsigned short* attnb = (unsigned short*)(base + off); off += (size_t)NCELL * EE * 2;

    (void)hipMemsetAsync(counts, 0, (size_t)NCELL * 4, stream);
    bin_kernel<<<NPTS / 256, 256, 0, stream>>>(x, counts, cellPts);

    // Precision staging: bf16 activations, transposed bf16 weights
    convert_bf16_kernel<<<(NPTS * EE) / 2048, 256, 0, stream>>>(z, zb);
    convert_bf16_kernel<<<(SS * EE) / 2048, 256, 0, stream>>>(lat, latb);
    transpose_w_kernel<<<EE, EE, 0, stream>>>(wq, wqT);
    transpose_w_kernel<<<EE, EE, 0, stream>>>(wk, wkT);
    transpose_w_kernel<<<EE, EE, 0, stream>>>(wv, wvT);
    transpose_w_kernel<<<EE, EE, 0, stream>>>(wo, woT);

    // Point projections: [131072,256] @ [256,256]
    dim3 gP(2, NPTS / 128);
    gemm_bf16_wmma<<<gP, 256, 0, stream>>>(zb, wkT, k_pts, NPTS);
    gemm_bf16_wmma<<<gP, 256, 0, stream>>>(zb, wvT, v_pts, NPTS);

    // Latent projections: [1024,256] @ [256,256]
    dim3 gL(2, SS / 128);
    gemm_bf16_wmma<<<gL, 256, 0, stream>>>(latb, wqT, q_lat, SS);
    gemm_bf16_wmma<<<gL, 256, 0, stream>>>(latb, wkT, k_lat, SS);
    gemm_bf16_wmma<<<gL, 256, 0, stream>>>(latb, wvT, v_lat, SS);

    // Per-cell attention (one wave per cell), bf16 output
    attn_kernel<<<NCELL / 8, 256, 0, stream>>>(q_lat, k_lat, v_lat, k_pts, v_pts,
                                               counts, cellPts, attnb);

    // Output projection straight into d_out's z_grid region
    dim3 gO(2, NCELL / 128);
    gemm_bf16_wmma<<<gO, 256, 0, stream>>>(attnb, woT, out + (size_t)BB * SS * 2, NCELL);

    // x_grid broadcast
    xgrid_kernel<<<(BB * SS * 2) / 256, 256, 0, stream>>>(grid, out);
}
